// RPN_65128884077255
// MI455X (gfx1250) — matrix operand order repaired and checked by reading, not compiled, not run on hardware
//
#include <hip/hip_runtime.h>
#include <stdint.h>

// ---------------------------------------------------------------------------
// CDNA5 / gfx1250 types
// ---------------------------------------------------------------------------
typedef __attribute__((ext_vector_type(16))) __bf16 bf16x16;
typedef __attribute__((ext_vector_type(8)))  __bf16 bf16x8;
typedef __attribute__((ext_vector_type(8)))  float  f32x8;

static __device__ __forceinline__ unsigned f2bf_bits(float f) {
  // round-to-nearest-even fp32 -> bf16 (bits in low 16)
  unsigned u = __float_as_uint(f);
  return (u + 0x7FFFu + ((u >> 16) & 1u)) >> 16;
}
static __device__ __forceinline__ __bf16 f2bf(float f) {
  unsigned short s = (unsigned short)f2bf_bits(f);
  return __builtin_bit_cast(__bf16, s);
}

// ---------------------------------------------------------------------------
// Kernel 1: reorder conv weights (O, Cin, 3,3,3) f32 -> bf16 [O][tap*Cin+ci]
// so that a 32-wide K tile has a constant filter tap (cheap implicit GEMM).
// ---------------------------------------------------------------------------
__global__ void pack_weights(const float* __restrict__ w, __bf16* __restrict__ out,
                             int O, int Cin) {
  int total = O * Cin * 27;
  int idx = blockIdx.x * blockDim.x + threadIdx.x;
  if (idx >= total) return;
  int o   = idx / (Cin * 27);
  int r   = idx - o * (Cin * 27);
  int ci  = r / 27;
  int tap = r - ci * 27;
  out[(size_t)o * (Cin * 27) + (size_t)tap * Cin + ci] = f2bf(w[idx]);
}

// ---------------------------------------------------------------------------
// Kernel 2: 3D conv (kernel 3x3x3, pad 1, stride (1,2,2)) as implicit GEMM
// using v_wmma_f32_16x16x32_bf16.  M=512 out channels, N=T*Hout*Wout,
// K=27*Cin (K order = tap*Cin+ci so tap is constant per 32-wide K tile).
// WG: 256 threads (8 waves), tile 128(M) x 128(N), K step 32.
// Waves: 4(M) x 2(N); each wave owns 32(M) x 64(N) = 2x4 WMMA tiles.
// A tile staged global->LDS with GLOBAL_LOAD_ASYNC_TO_LDS_B128 (ASYNCcnt).
// ---------------------------------------------------------------------------
#define MT  128
#define NT  128
#define KT  32
#define LDA 40   // bf16 elems per A row in LDS (80B, 16B aligned, padded)
#define LDB 40   // bf16 elems per B row in LDS

__global__ __launch_bounds__(256)
void conv3d_wmma(const float* __restrict__ in, const __bf16* __restrict__ Wp,
                 const float* __restrict__ bias, float* __restrict__ out,
                 int Cin, int Tt, int Hin, int Win, int Hout, int Wout) {
  __shared__ __attribute__((aligned(16))) __bf16 As[MT * LDA];
  __shared__ __attribute__((aligned(16))) __bf16 Bs[NT * LDB];

  const int tid   = threadIdx.x;
  const int lane  = tid & 31;
  const int wv    = tid >> 5;        // wave 0..7
  const int wmW   = (wv & 3) * 32;   // wave M offset in WG tile
  const int wnW   = (wv >> 2) * 64;  // wave N offset in WG tile
  const int laneM = lane & 15;
  const int laneH = lane >> 4;       // 0 | 1

  const int Nn    = Tt * Hout * Wout;
  const int KD    = 27 * Cin;
  const int mBase = blockIdx.y * MT;
  const int nBase = blockIdx.x * NT;

  // --- B (input patch) mapping: thread -> (n, contiguous 16-wide kk run) --
  const int nLoc   = tid & 127;
  const int kkBase = (tid >> 7) * 16;              // 0 or 16
  int p   = nBase + nLoc;
  int t   = p / (Hout * Wout);
  int rem = p - t * (Hout * Wout);
  int h   = rem / Wout;
  int w   = rem - h * Wout;

  // --- A (weights) mapping: thread copies one row-chunk of 32B -----------
  const int ar  = tid >> 1;          // row 0..127
  const int akk = (tid & 1) * 16;    // 0 or 16
  const __bf16* wrow = Wp + (size_t)(mBase + ar) * KD + akk;
  const uint64_t gaBase = (uint64_t)(uintptr_t)wrow;
  const unsigned ldsA   = (unsigned)(uintptr_t)&As[ar * LDA + akk];

  const long ciStride = (long)Tt * Hin * Win;

  f32x8 acc[2][4];
#pragma unroll
  for (int ms = 0; ms < 2; ++ms)
#pragma unroll
    for (int ns = 0; ns < 4; ++ns)
#pragma unroll
      for (int e = 0; e < 8; ++e) acc[ms][ns][e] = 0.0f;

  for (int k0 = 0; k0 < KD; k0 += KT) {
    const int tap = k0 / Cin;               // constant tap within the tile
    const int ciB = k0 - tap * Cin;
    const int dz  = tap / 9;
    const int ry  = tap - dz * 9;
    const int dy  = ry / 3;
    const int dx  = ry - dy * 3;

    // A: async DMA global -> LDS (two 16B chunks; IOFFSET applies to both
    // the LDS and the global address).  Tracked by ASYNCcnt.
    {
      const uint64_t ga = gaBase + ((uint64_t)k0 << 1);
      asm volatile("global_load_async_to_lds_b128 %0, %1, off\n\t"
                   "global_load_async_to_lds_b128 %0, %1, off offset:16"
                   :: "v"(ldsA), "v"(ga) : "memory");
    }
    __builtin_prefetch(wrow + k0 + KT, 0, 3);   // global_prefetch_b8 next tile

    // B: gather f32 input patch, convert bf16, pack pairs, 2x ds_store_b128
    const int it = t + dz - 1;
    const int ih = h * 2 + dy - 1;
    const int iw = w * 2 + dx - 1;
    const bool valid = ((unsigned)it < (unsigned)Tt) &
                       ((unsigned)ih < (unsigned)Hin) &
                       ((unsigned)iw < (unsigned)Win);
    const float* src = in + ((long)it * Hin + ih) * Win + iw +
                       (long)ciB * ciStride;
    float v[16];
#pragma unroll
    for (int i = 0; i < 16; ++i) {
      v[i] = 0.0f;
      if (valid) v[i] = src[(long)(kkBase + i) * ciStride];
    }
    uint4 pk0, pk1;
    pk0.x = f2bf_bits(v[0])  | (f2bf_bits(v[1])  << 16);
    pk0.y = f2bf_bits(v[2])  | (f2bf_bits(v[3])  << 16);
    pk0.z = f2bf_bits(v[4])  | (f2bf_bits(v[5])  << 16);
    pk0.w = f2bf_bits(v[6])  | (f2bf_bits(v[7])  << 16);
    pk1.x = f2bf_bits(v[8])  | (f2bf_bits(v[9])  << 16);
    pk1.y = f2bf_bits(v[10]) | (f2bf_bits(v[11]) << 16);
    pk1.z = f2bf_bits(v[12]) | (f2bf_bits(v[13]) << 16);
    pk1.w = f2bf_bits(v[14]) | (f2bf_bits(v[15]) << 16);
    *(uint4*)&Bs[nLoc * LDB + kkBase]     = pk0;
    *(uint4*)&Bs[nLoc * LDB + kkBase + 8] = pk1;

    asm volatile("s_wait_asynccnt 0x0" ::: "memory");
    __syncthreads();

    // Fragments straight from LDS in the hardware register layout.
    bf16x16 aF[2], bF[4];
#pragma unroll
    for (int ms = 0; ms < 2; ++ms) {
      // A 16-bit 16x32: lanes0-15 -> K 0-7 & 16-23 ; lanes16-31 -> 8-15 & 24-31
      const __bf16* ap = &As[(wmW + ms * 16 + laneM) * LDA + laneH * 8];
      bf16x8 lo = *(const bf16x8*)ap;
      bf16x8 hi = *(const bf16x8*)(ap + 16);
#pragma unroll
      for (int e = 0; e < 8; ++e) { aF[ms][e] = lo[e]; aF[ms][8 + e] = hi[e]; }
    }
#pragma unroll
    for (int ns = 0; ns < 4; ++ns) {
      // B 16-bit 32x16: lane holds column N=lane&15; lanes0-15 -> K 0-7 & 16-23 ; lanes16-31 -> 8-15 & 24-31
      const __bf16* bp = &Bs[(wnW + ns * 16 + laneM) * LDB + laneH * 8];
      bf16x8 lo = *(const bf16x8*)bp;
      bf16x8 hi = *(const bf16x8*)(bp + 16);
#pragma unroll
      for (int e = 0; e < 8; ++e) { bF[ns][e] = lo[e]; bF[ns][8 + e] = hi[e]; }
    }

#pragma unroll
    for (int ms = 0; ms < 2; ++ms)
#pragma unroll
      for (int ns = 0; ns < 4; ++ns)
        acc[ms][ns] = __builtin_amdgcn_wmma_f32_16x16x32_bf16(
            false, aF[ms], false, bF[ns], (short)0, acc[ms][ns], false, false);
    __syncthreads();
  }

  // Epilogue: C/D layout -> VGPR j holds M=j+8*laneH, N=laneM. Fuse bias.
#pragma unroll
  for (int ms = 0; ms < 2; ++ms) {
#pragma unroll
    for (int ns = 0; ns < 4; ++ns) {
      const int col = nBase + wnW + ns * 16 + laneM;
#pragma unroll
      for (int j = 0; j < 8; ++j) {
        const int row = mBase + wmW + ms * 16 + j + laneH * 8;
        out[(size_t)row * Nn + col] = acc[ms][ns][j] + bias[row];
      }
    }
  }
}

// ---------------------------------------------------------------------------
// Kernel 3: max over H*W -> feat[c][t]
// ---------------------------------------------------------------------------
__global__ void maxpool_hw(const float* __restrict__ x, float* __restrict__ feat,
                           int Cc, int Tt, int Ss) {
  int idx = blockIdx.x * blockDim.x + threadIdx.x;
  if (idx >= Cc * Tt) return;
  int c = idx / Tt, t = idx - c * Tt;
  const float* p = x + ((size_t)c * Tt + t) * Ss;
  float m = p[0];
  for (int s = 1; s < Ss; ++s) m = fmaxf(m, p[s]);
  feat[c * Tt + t] = m;
}

// ---------------------------------------------------------------------------
// Kernel 4: score/delta heads: (16x512) x feat(512x64) + bias (f32, tiny)
// ---------------------------------------------------------------------------
__global__ void heads_kernel(const float* __restrict__ feat,
                             const float* __restrict__ sw, const float* __restrict__ sb,
                             const float* __restrict__ dw, const float* __restrict__ db,
                             float* sOut, float* dOut) {
  int idx = blockIdx.x * blockDim.x + threadIdx.x;  // 0..2047
  if (idx >= 2 * 16 * 64) return;
  int hid = idx >> 10;
  int r   = idx & 1023;
  int o   = r >> 6, t = r & 63;
  const float* wv = (hid ? dw : sw) + o * 512;
  float a = (hid ? db : sb)[o];
  for (int c = 0; c < 512; ++c) a += wv[c] * feat[c * 64 + t];
  (hid ? dOut : sOut)[o * 64 + t] = a;
}

// ---------------------------------------------------------------------------
// Kernel 5: softmax, anchor/box decode, NMS(100), labels, losses.
// One block of 512 threads, thread n == anchor n (n = a*64 + t).
// ---------------------------------------------------------------------------
__global__ __launch_bounds__(512)
void finalize(const float* __restrict__ S, const float* __restrict__ D,
              const float* __restrict__ gt, const int* __restrict__ vlen,
              float* __restrict__ out) {
  __shared__ float sBs[512], sBe[512], sSc[512];
  __shared__ float rv[512];
  __shared__ int   ri[512];
  __shared__ float sIou[8][512];
  __shared__ int   sBest[8];

  const int n = threadIdx.x;
  const int a = n >> 6, t = n & 63;
  const float scales[8] = {1.f, 2.f, 4.f, 8.f, 16.f, 32.f, 64.f, 128.f};
  const float alen   = 8.0f * scales[a];
  const float center = (t + 0.5f) * 8.0f;
  const float lim    = 512.0f;

  // log-softmax over 2 classes; logits at S[n] (cls0), S[512+n] (cls1)
  float l0 = S[n], l1 = S[512 + n];
  float mx  = fmaxf(l0, l1);
  float lse = mx + __logf(__expf(l0 - mx) + __expf(l1 - mx));
  float lp0 = l0 - lse, lp1 = l1 - lse;
  float p1  = __expf(lp1);

  // box decode
  float dc = D[n], dl = D[512 + n];
  float pc = center + dc * alen;
  float pl = alen * __expf(fminf(fmaxf(dl, -10.f), 10.f));
  float ps = fminf(fmaxf(pc - 0.5f * pl, 0.f), lim);
  float pe = fminf(fmaxf(pc + 0.5f * pl, 0.f), lim);
  float fg = ((pe - ps) >= 4.0f) ? p1 : -1e9f;

  float as_ = center - 0.5f * alen;
  float ae_ = center + 0.5f * alen;

  sBs[n] = ps; sBe[n] = pe; sSc[n] = fg;
  __syncthreads();

  // ---- NMS: 100 sequential picks, parallel argmax + suppression ----------
  for (int it = 0; it < 100; ++it) {
    rv[n] = sSc[n]; ri[n] = n;
    __syncthreads();
    for (int s = 256; s > 0; s >>= 1) {
      if (n < s) {
        float v2 = rv[n + s]; int i2 = ri[n + s];
        if (v2 > rv[n] || (v2 == rv[n] && i2 < ri[n])) { rv[n] = v2; ri[n] = i2; }
      }
      __syncthreads();
    }
    int   bi  = ri[0];
    float bs  = rv[0];
    float bx0 = sBs[bi], bx1 = sBe[bi];

    float inter = fmaxf(fminf(bx1, sBe[n]) - fmaxf(bx0, sBs[n]), 0.f);
    float uni   = (bx1 - bx0) + (sBe[n] - sBs[n]) - inter;
    float iou   = inter / fmaxf(uni, 1e-6f);
    if (iou > 0.7f || n == bi) sSc[n] = -1e9f;

    if (n == 0) {
      bool keep = bs > -5e8f;
      out[it * 2]     = keep ? bx0 : 0.f;
      out[it * 2 + 1] = keep ? bx1 : 0.f;
      out[200 + it]   = keep ? bs  : 0.f;
    }
    __syncthreads();
  }

  // ---- anchor-vs-GT IoU, labels -------------------------------------------
  float maxIou = -1.f; int argG = 0;
#pragma unroll
  for (int g = 0; g < 8; ++g) {
    float gs = gt[g * 2], ge = gt[g * 2 + 1];
    float inter = fmaxf(fminf(ae_, ge) - fmaxf(as_, gs), 0.f);
    float uni   = (ae_ - as_) + (ge - gs) - inter;
    float iou   = inter / fmaxf(uni, 1e-6f);
    sIou[g][n] = iou;
    if (iou > maxIou) { maxIou = iou; argG = g; }
  }
  __syncthreads();
  if (n < 8) {
    float bv = sIou[n][0]; int bidx = 0;
    for (int k = 1; k < 512; ++k)
      if (sIou[n][k] > bv) { bv = sIou[n][k]; bidx = k; }
    sBest[n] = bidx;
  }
  __syncthreads();

  int lab = -1;
  if (maxIou < 0.3f)  lab = 0;
  if (maxIou >= 0.7f) lab = 1;
#pragma unroll
  for (int g = 0; g < 8; ++g) if (sBest[g] == n) lab = 1;
  float vl = (float)vlen[0];
  if (!(as_ >= 0.f && ae_ <= vl)) lab = -1;

  // regression target from best-matching GT
  float gs = gt[argG * 2], ge = gt[argG * 2 + 1];
  float gc = 0.5f * (gs + ge);
  float gl = fmaxf(ge - gs, 1e-6f);
  float ac = 0.5f * (as_ + ae_);
  float al = ae_ - as_;
  float r0 = (gc - ac) / al;
  float r1 = __logf(gl / al);

  float mask   = (lab >= 0) ? 1.f : 0.f;
  float nll    = -((lab == 1) ? lp1 : lp0);
  float clsNum = nll * mask;

  float d0 = dc - r0, d1 = dl - r1;
  float s0 = fabsf(d0) < 1.f ? 0.5f * d0 * d0 : fabsf(d0) - 0.5f;
  float s1 = fabsf(d1) < 1.f ? 0.5f * d1 * d1 : fabsf(d1) - 0.5f;
  float pos    = (lab == 1) ? 1.f : 0.f;
  float regNum = (s0 + s1) * pos;

  float vals[4] = {clsNum, mask, regNum, pos};
  float sums[4];
  for (int q = 0; q < 4; ++q) {
    rv[n] = vals[q];
    __syncthreads();
    for (int s = 256; s > 0; s >>= 1) {
      if (n < s) rv[n] += rv[n + s];
      __syncthreads();
    }
    sums[q] = rv[0];
    __syncthreads();
  }
  if (n == 0) {
    out[300] = sums[0] / fmaxf(sums[1], 1.f);
    out[301] = sums[2] / fmaxf(sums[3], 1.f);
  }
}

// ---------------------------------------------------------------------------
// Host launcher
// ---------------------------------------------------------------------------
extern "C" void kernel_launch(void* const* d_in, const int* in_sizes, int n_in,
                              void* d_out, int out_size, void* d_ws, size_t ws_size,
                              hipStream_t stream) {
  (void)in_sizes; (void)n_in; (void)out_size; (void)ws_size;
  const float* base = (const float*)d_in[0];
  const float* gt   = (const float*)d_in[1];
  const int*   vlen = (const int*)d_in[2];
  const float* c1w  = (const float*)d_in[3];
  const float* c1b  = (const float*)d_in[4];
  const float* c2w  = (const float*)d_in[5];
  const float* c2b  = (const float*)d_in[6];
  const float* sw   = (const float*)d_in[7];
  const float* sb   = (const float*)d_in[8];
  const float* dw   = (const float*)d_in[9];
  const float* db   = (const float*)d_in[10];
  float* out = (float*)d_out;

  char* ws = (char*)d_ws;
  size_t off = 0;
  auto alloc = [&](size_t bytes) -> void* {
    void* p = ws + off;
    off = (off + bytes + 255) & ~(size_t)255;
    return p;
  };

  __bf16* wp1  = (__bf16*)alloc((size_t)512 * 6912 * 2);   // 7.0 MB
  __bf16* wp2  = (__bf16*)alloc((size_t)512 * 13824 * 2);  // 14.2 MB
  float*  o1   = (float*)alloc((size_t)512 * 16384 * 4);   // 33.6 MB
  float*  o2   = (float*)alloc((size_t)512 * 4096 * 4);    // 8.4 MB
  float*  feat = (float*)alloc((size_t)512 * 64 * 4);
  float*  sbuf = (float*)alloc((size_t)16 * 64 * 4);
  float*  dbuf = (float*)alloc((size_t)16 * 64 * 4);

  {
    int total = 512 * 256 * 27;
    pack_weights<<<(total + 255) / 256, 256, 0, stream>>>(c1w, wp1, 512, 256);
  }
  {
    int total = 512 * 512 * 27;
    pack_weights<<<(total + 255) / 256, 256, 0, stream>>>(c2w, wp2, 512, 512);
  }

  // conv1: Cin=256, in 64x32x32, out 512 x (64x16x16)
  conv3d_wmma<<<dim3(16384 / NT, 512 / MT), 256, 0, stream>>>(
      base, wp1, c1b, o1, 256, 64, 32, 32, 16, 16);
  // conv2: Cin=512, in 64x16x16, out 512 x (64x8x8)
  conv3d_wmma<<<dim3(4096 / NT, 512 / MT), 256, 0, stream>>>(
      o1, wp2, c2b, o2, 512, 64, 16, 16, 8, 8);

  maxpool_hw<<<(512 * 64 + 255) / 256, 256, 0, stream>>>(o2, feat, 512, 64, 64);
  heads_kernel<<<(2048 + 255) / 256, 256, 0, stream>>>(feat, sw, sb, dw, db, sbuf, dbuf);
  finalize<<<1, 512, 0, stream>>>(sbuf, dbuf, gt, vlen, out);
}
